// LocalizedPromptedAttentionLayer2_46110768890014
// MI455X (gfx1250) — matrix operand order repaired and checked by reading, not compiled, not run on hardware
//
#include <hip/hip_runtime.h>
#include <math.h>

// Problem constants (from reference)
#define B_   8
#define NX_  512
#define W_   16
#define DX_  1536
#define DY_  1024
#define D_   768
#define H_   12
#define P_   8
#define DH_  64
#define NT_  (D_/16)   // 48 column tiles of 16
#define WPB_ 2         // windows per block (M = 32 rows) -> halves L2 weight traffic

typedef __attribute__((ext_vector_type(16))) _Float16 v16h;
typedef __attribute__((ext_vector_type(8)))  _Float16 v8h;
typedef __attribute__((ext_vector_type(8)))  float    v8f;

__device__ __forceinline__ v8f wmma_f16(v16h a, v16h b, v8f c) {
  // D = A(16x32 f16) * B(32x16 f16) + C(16x16 f32)
  return __builtin_amdgcn_wmma_f32_16x16x32_f16(false, a, false, b, (short)0, c,
                                                false, false);
}

__device__ __forceinline__ v16h combine8(v8h a0, v8h a1) {
  v16h r;
#pragma unroll
  for (int i = 0; i < 8; ++i) { r[i] = a0[i]; r[8 + i] = a1[i]; }
  return r;
}

// A fragment (16x32 f16) from an LDS row-major f16 matrix with row stride D_.
// ISA layout: lane m = lane&15; K = 8*hf + [0..7] (v0..3),
// K = 16 + 8*hf + [0..7] (v4..7)  ->  two contiguous 16B chunks.
__device__ __forceinline__ v16h load_a_lds(const _Float16* m, int row, int k0, int hf) {
  const v8h* p0 = (const v8h*)(m + (size_t)row * D_ + k0 + 8 * hf);
  const v8h* p1 = (const v8h*)(m + (size_t)row * D_ + k0 + 16 + 8 * hf);
  return combine8(*p0, *p1);
}

// B fragment (32x16 f16) from the pre-packed weight buffer:
// packed[((kt*NT_+nt)*32 + lane)*16 + e], K_local = 8*hf + e (e<8), 16 + 8*hf + (e-8) (e>=8).
__device__ __forceinline__ v16h load_b_pk(const _Float16* pk, int kt, int nt, int lane) {
  const v8h* p = (const v8h*)(pk + (((size_t)kt * NT_ + nt) * 32 + (size_t)lane) * 16);
  return combine8(p[0], p[1]);
}

// A fragment from global fp32 row (cvt to f16 in regs).
__device__ __forceinline__ v16h load_a_global(const float* yr, int k0, int hf) {
  const float4* g0 = (const float4*)(yr + k0 + 8 * hf);
  const float4* g1 = (const float4*)(yr + k0 + 16 + 8 * hf);
  float4 f0 = g0[0], f1 = g0[1], f2 = g1[0], f3 = g1[1];
  v16h A;
  A[0]  = (_Float16)f0.x; A[1]  = (_Float16)f0.y; A[2]  = (_Float16)f0.z; A[3]  = (_Float16)f0.w;
  A[4]  = (_Float16)f1.x; A[5]  = (_Float16)f1.y; A[6]  = (_Float16)f1.z; A[7]  = (_Float16)f1.w;
  A[8]  = (_Float16)f2.x; A[9]  = (_Float16)f2.y; A[10] = (_Float16)f2.z; A[11] = (_Float16)f2.w;
  A[12] = (_Float16)f3.x; A[13] = (_Float16)f3.y; A[14] = (_Float16)f3.z; A[15] = (_Float16)f3.w;
  return A;
}

// ---------------------------------------------------------------------------
// Pack fp32 weight [K][768] into WMMA B-fragment order as f16.
// Grid: (K/32)*NT_ blocks, 32 threads (one wave) per block.
// ---------------------------------------------------------------------------
__global__ void pack_w_kernel(const float* __restrict__ W, _Float16* __restrict__ out) {
  int blk  = blockIdx.x;            // kt*NT_ + nt
  int kt   = blk / NT_;
  int nt   = blk - kt * NT_;
  int lane = threadIdx.x;           // 0..31
  int n    = lane & 15;
  int hf   = lane >> 4;
  _Float16* dst = out + ((size_t)blk * 32 + lane) * 16;
  const float* src = W + (size_t)(kt * 32 + 8 * hf) * D_ + nt * 16 + n;
#pragma unroll
  for (int e = 0; e < 16; ++e) dst[e] = (_Float16)src[(size_t)(e < 8 ? e : e + 8) * D_];
}

// ---------------------------------------------------------------------------
// q = l2n(x[:,0] @ Wx + bx) @ Wq + bq, scaled by 1/sqrt(dh).
// Only row 0 of x contributes to the reference output. Grid: B_ x 256.
// ---------------------------------------------------------------------------
__global__ void q_kernel(const float* __restrict__ x, const float* __restrict__ Wx,
                         const float* __restrict__ bx, const float* __restrict__ Wq,
                         const float* __restrict__ bq, float* __restrict__ q_ws) {
  __shared__ float xs[DX_];
  __shared__ float xn[D_];
  __shared__ float red[256];
  int b = blockIdx.x, tid = threadIdx.x;
  const float* xr = x + (size_t)b * NX_ * DX_;   // x[b, 0, :]
  for (int j = 0; j < DX_ / 256; ++j) xs[tid + 256 * j] = xr[tid + 256 * j];
  __syncthreads();
  float psq = 0.f;
  for (int j = 0; j < 3; ++j) {
    int d = tid + 256 * j;
    float acc = bx[d];
    for (int k = 0; k < DX_; ++k) acc = fmaf(xs[k], Wx[(size_t)k * D_ + d], acc);
    xn[d] = acc;
    psq += acc * acc;
  }
  red[tid] = psq;
  __syncthreads();
  for (int s = 128; s > 0; s >>= 1) {
    if (tid < s) red[tid] += red[tid + s];
    __syncthreads();
  }
  float inv = 1.0f / (sqrtf(red[0]) + 1e-6f);
  __syncthreads();
  for (int j = 0; j < 3; ++j) xn[tid + 256 * j] *= inv;
  __syncthreads();
  for (int j = 0; j < 3; ++j) {
    int d = tid + 256 * j;
    float acc = bq[d];
    for (int c = 0; c < D_; ++c) acc = fmaf(xn[c], Wq[(size_t)c * D_ + d], acc);
    q_ws[(size_t)b * D_ + d] = acc * 0.125f;   // 1/sqrt(64)
  }
}

// ---------------------------------------------------------------------------
// Fused main kernel: one workgroup per PAIR of 16-token windows (M = 32).
// 256 threads = 8 wave32; each B fragment feeds 2 A fragments (one/window).
// ---------------------------------------------------------------------------
__global__ __launch_bounds__(256)
void attn_main_kernel(const float* __restrict__ y, const float* __restrict__ by,
                      const _Float16* __restrict__ pWy, const _Float16* __restrict__ pWk,
                      const _Float16* __restrict__ pWv, const _Float16* __restrict__ pWo,
                      const float* __restrict__ bk, const float* __restrict__ bv,
                      const float* __restrict__ bo,
                      const float* __restrict__ ln_g, const float* __restrict__ ln_b,
                      const float* __restrict__ prompt, const float* __restrict__ q_ws,
                      float* __restrict__ out) {
  __shared__ __align__(16) _Float16 sh_ypH[32 * D_];   // 48 KB: l2-normed yp (2 windows)
  __shared__ __align__(16) _Float16 sh_kmr[32 * D_];   // 48 KB: k matrix, reused as rbar
  __shared__ __align__(16) _Float16 sh_ctxh[WPB_ * D_];// ctx vectors (f16) for Wo GEMM
  __shared__ float sh_qv[D_];
  __shared__ float sh_sc[WPB_ * 192];
  __shared__ float sh_ov[WPB_ * D_];
  __shared__ float sh_red[256];
  __shared__ float sh_red2[256];
  __shared__ float sh_rowsq[32];

  const int tid  = threadIdx.x;
  const int wave = tid >> 5;
  const int lane = tid & 31;
  const int ln15 = lane & 15;
  const int hf   = lane >> 4;
  const int b    = blockIdx.x >> 8;    // / (NX_/WPB_)
  const int np   = blockIdx.x & 255;   // window-pair index

  // ---- phase 0: stage q, zero row-sumsq -----------------------------------
  for (int j = 0; j < 3; ++j) sh_qv[tid + 256 * j] = q_ws[(size_t)b * D_ + tid + 256 * j];
  if (tid < 32) sh_rowsq[tid] = 0.f;
  __syncthreads();

  // ---- phase 1: yp_raw = y_pair(32x1024) @ Wy + by  (WMMA, K=1024) --------
  // B loaded once per kt, reused by both windows' A fragments.
  const float* ybase = y + ((size_t)b * (NX_ * W_) + (size_t)np * (WPB_ * W_)) * DY_;
  {
    v8f accs[WPB_][6] = {};
#pragma unroll 1
    for (int kt = 0; kt < DY_ / 32; ++kt) {
      const int k0 = kt * 32;
      v16h Aw[WPB_];
#pragma unroll
      for (int w = 0; w < WPB_; ++w)
        Aw[w] = load_a_global(ybase + (size_t)(w * 16 + ln15) * DY_, k0, hf);
#pragma unroll
      for (int t = 0; t < 6; ++t) {
        v16h Bv = load_b_pk(pWy, kt, wave * 6 + t, lane);
#pragma unroll
        for (int w = 0; w < WPB_; ++w) accs[w][t] = wmma_f16(Aw[w], Bv, accs[w][t]);
      }
    }
    float sq[WPB_][8];
#pragma unroll
    for (int w = 0; w < WPB_; ++w)
#pragma unroll
      for (int r = 0; r < 8; ++r) sq[w][r] = 0.f;
#pragma unroll
    for (int w = 0; w < WPB_; ++w)
#pragma unroll
      for (int t = 0; t < 6; ++t) {
        const int n0 = (wave * 6 + t) * 16;
        const float bias = by[n0 + ln15];
#pragma unroll
        for (int r = 0; r < 8; ++r) {
          float v = accs[w][t][r] + bias;
          sq[w][r] += v * v;
          sh_ypH[(size_t)(w * 16 + r + 8 * hf) * D_ + n0 + ln15] = (_Float16)v;
        }
      }
    // row sum-of-squares: reduce across the 16 lanes sharing each output row
#pragma unroll
    for (int w = 0; w < WPB_; ++w)
#pragma unroll
      for (int r = 0; r < 8; ++r) {
        float v = sq[w][r];
        v += __shfl_xor(v, 1); v += __shfl_xor(v, 2);
        v += __shfl_xor(v, 4); v += __shfl_xor(v, 8);
        if (ln15 == 0) atomicAdd(&sh_rowsq[w * 16 + 8 * hf + r], v);
      }
  }
  __syncthreads();

  // ---- phase 2: l2 normalize the 32 yp rows in place (f16) ----------------
  if (tid < 32) sh_rowsq[tid] = 1.0f / (sqrtf(sh_rowsq[tid]) + 1e-6f);
  __syncthreads();
  {
    const int row = tid >> 3;
    const float rn = sh_rowsq[row];
    const int c0 = (tid & 7) * 96;
    for (int i = 0; i < 96; ++i) {
      const int idx = row * D_ + c0 + i;
      sh_ypH[idx] = (_Float16)((float)sh_ypH[idx] * rn);
    }
  }
  __syncthreads();

  // ---- phase 3: k = ypH @ Wk + bk  (WMMA, K=768, both windows) ------------
  {
    v8f accs[WPB_][6] = {};
#pragma unroll 1
    for (int kt = 0; kt < D_ / 32; ++kt) {
      v16h Aw[WPB_];
#pragma unroll
      for (int w = 0; w < WPB_; ++w)
        Aw[w] = load_a_lds(sh_ypH, w * 16 + ln15, kt * 32, hf);
#pragma unroll
      for (int t = 0; t < 6; ++t) {
        v16h Bv = load_b_pk(pWk, kt, wave * 6 + t, lane);
#pragma unroll
        for (int w = 0; w < WPB_; ++w) accs[w][t] = wmma_f16(Aw[w], Bv, accs[w][t]);
      }
    }
#pragma unroll
    for (int w = 0; w < WPB_; ++w)
#pragma unroll
      for (int t = 0; t < 6; ++t) {
        const int n0 = (wave * 6 + t) * 16;
        const float bias = bk[n0 + ln15];
#pragma unroll
        for (int r = 0; r < 8; ++r)
          sh_kmr[(size_t)(w * 16 + r + 8 * hf) * D_ + n0 + ln15] =
              (_Float16)(accs[w][t][r] + bias);
      }
  }
  __syncthreads();

  // ---- phase 4: scores s[win,h,w] = q[h]·k[win,w,h] ; softmax over w ------
  for (int win = 0; win < WPB_; ++win) {
    if (tid < H_ * 16) {
      const int h = tid >> 4, w = tid & 15;
      const _Float16* krow = sh_kmr + (size_t)(win * 16 + w) * D_ + h * DH_;
      float s = 0.f;
      for (int d = 0; d < DH_; ++d) s += sh_qv[h * DH_ + d] * (float)krow[d];
      sh_sc[win * 192 + tid] = s;
    }
  }
  __syncthreads();
  if (tid < WPB_ * H_) {
    const int win = (tid >= H_) ? 1 : 0;
    const int h   = tid - win * H_;
    float* sc = sh_sc + win * 192 + h * 16;
    float mx = -1e30f;
    for (int w = 0; w < 16; ++w) mx = fmaxf(mx, sc[w]);
    float e[16], sum = 0.f;
    for (int w = 0; w < 16; ++w) { e[w] = expf(sc[w] - mx); sum += e[w]; }
    const float invs = 1.0f / sum;
    for (int w = 0; w < 16; ++w) sc[w] = e[w] * invs;
  }
  __syncthreads();

  // ---- phase 5: rbar[win*16+h,:] = sum_w a[win,h,w] * ypH[win*16+w,:] -----
  //      (attention is linear, so ctx = rbar @ Wv + bv). Overwrites sh_kmr.
  for (int row = 0; row < 32; ++row) {
    const int win = row >> 4, h = row & 15;
#pragma unroll
    for (int j = 0; j < 3; ++j) {
      const int c = tid + 256 * j;
      float v = 0.f;
      if (h < H_) {
        for (int w = 0; w < W_; ++w)
          v += sh_sc[win * 192 + h * 16 + w] * (float)sh_ypH[(size_t)(win * 16 + w) * D_ + c];
      }
      sh_kmr[(size_t)row * D_ + c] = (_Float16)v;
    }
  }
  __syncthreads();

  // ---- phase 6: ctx = rbar @ Wv + bv (WMMA); keep head-diagonal blocks ----
  {
    v8f accs[WPB_][6] = {};
#pragma unroll 1
    for (int kt = 0; kt < D_ / 32; ++kt) {
      v16h Aw[WPB_];
#pragma unroll
      for (int w = 0; w < WPB_; ++w)
        Aw[w] = load_a_lds(sh_kmr, w * 16 + ln15, kt * 32, hf);
#pragma unroll
      for (int t = 0; t < 6; ++t) {
        v16h Bv = load_b_pk(pWv, kt, wave * 6 + t, lane);
#pragma unroll
        for (int w = 0; w < WPB_; ++w) accs[w][t] = wmma_f16(Aw[w], Bv, accs[w][t]);
      }
    }
#pragma unroll
    for (int w = 0; w < WPB_; ++w)
#pragma unroll
      for (int t = 0; t < 6; ++t) {
        const int col = (wave * 6 + t) * 16 + ln15;
#pragma unroll
        for (int r = 0; r < 8; ++r) {
          if ((r + 8 * hf) == (col >> 6))
            sh_ctxh[w * D_ + col] = (_Float16)(accs[w][t][r] + bv[col]);
        }
      }
  }
  __syncthreads();

  // ---- phase 7: o = ctx @ Wo + bo via WMMA (A row 0 = ctx, rows 1..15 = 0) -
  {
    v8f accs[WPB_][6] = {};
#pragma unroll 1
    for (int kt = 0; kt < D_ / 32; ++kt) {
      v16h Aw[WPB_];
#pragma unroll
      for (int w = 0; w < WPB_; ++w) {
        v16h A;
#pragma unroll
        for (int i = 0; i < 16; ++i) A[i] = (_Float16)0.0f;
        if (ln15 == 0) A = load_a_lds(sh_ctxh + (size_t)w * D_, 0, kt * 32, hf);
        Aw[w] = A;
      }
#pragma unroll
      for (int t = 0; t < 6; ++t) {
        v16h Bv = load_b_pk(pWo, kt, wave * 6 + t, lane);
#pragma unroll
        for (int w = 0; w < WPB_; ++w) accs[w][t] = wmma_f16(Aw[w], Bv, accs[w][t]);
      }
    }
    // output row 0 lives in acc[0] of lanes 0..15 (hf==0)
    if (lane < 16) {
#pragma unroll
      for (int w = 0; w < WPB_; ++w)
#pragma unroll
        for (int t = 0; t < 6; ++t) {
          const int col = (wave * 6 + t) * 16 + lane;
          sh_ov[w * D_ + col] = accs[w][t][0] + bo[col];
        }
    }
  }
  __syncthreads();

  // ---- phase 8: LayerNorm per window ; write [normed | prompt×P] ----------
  for (int win = 0; win < WPB_; ++win) {
    float oloc[3];
    float s1 = 0.f, s2 = 0.f;
#pragma unroll
    for (int j = 0; j < 3; ++j) {
      const float v = sh_ov[win * D_ + tid + 256 * j];
      oloc[j] = v;
      s1 += v;
      s2 += v * v;
    }
    sh_red[tid] = s1; sh_red2[tid] = s2;
    __syncthreads();
    for (int s = 128; s > 0; s >>= 1) {
      if (tid < s) { sh_red[tid] += sh_red[tid + s]; sh_red2[tid] += sh_red2[tid + s]; }
      __syncthreads();
    }
    const float mean = sh_red[0] * (1.0f / D_);
    const float var  = sh_red2[0] * (1.0f / D_) - mean * mean;
    const float rstd = rsqrtf(var + 1e-6f);

    const int n = np * WPB_ + win;
    const size_t obase = ((size_t)(b * NX_ + n) * (1 + P_)) * D_;
#pragma unroll
    for (int j = 0; j < 3; ++j) {
      const int dout = tid + 256 * j;
      out[obase + dout] = (oloc[j] - mean) * rstd * ln_g[dout] + ln_b[dout];
    }
    for (int p = 0; p < P_; ++p) {
#pragma unroll
      for (int j = 0; j < 3; ++j) {
        const int d = tid + 256 * j;
        out[obase + (size_t)(p + 1) * D_ + d] = prompt[p * D_ + d];
      }
    }
    __syncthreads();   // protect sh_red reuse by next window
  }
}

// ---------------------------------------------------------------------------
extern "C" void kernel_launch(void* const* d_in, const int* in_sizes, int n_in,
                              void* d_out, int out_size, void* d_ws, size_t ws_size,
                              hipStream_t stream) {
  const float* x      = (const float*)d_in[0];
  const float* y      = (const float*)d_in[1];
  const float* Wx     = (const float*)d_in[2];
  const float* bx     = (const float*)d_in[3];
  const float* Wy     = (const float*)d_in[4];
  const float* by     = (const float*)d_in[5];
  const float* prompt = (const float*)d_in[6];
  const float* Wq     = (const float*)d_in[7];
  const float* bq     = (const float*)d_in[8];
  const float* Wk     = (const float*)d_in[9];
  const float* bk     = (const float*)d_in[10];
  const float* Wv     = (const float*)d_in[11];
  const float* bv     = (const float*)d_in[12];
  const float* Wo     = (const float*)d_in[13];
  const float* bo     = (const float*)d_in[14];
  const float* ln_g   = (const float*)d_in[15];
  const float* ln_b   = (const float*)d_in[16];
  float* out = (float*)d_out;

  // Workspace layout (bytes):
  //   [0, 24576)          q_ws : f32[8][768]
  //   [32768, +1572864)   pWy  : f16 packed, (1024/32)*48 tiles * 512 halves
  //   [.., +1179648)      pWk  : f16 packed, (768/32)*48 tiles
  //   [.., +1179648)      pWv  : f16 packed
  //   [.., +1179648)      pWo  : f16 packed
  char* ws = (char*)d_ws;
  float*    q_ws = (float*)ws;
  _Float16* pWy  = (_Float16*)(ws + 32768);
  _Float16* pWk  = (_Float16*)(ws + 32768 + 1572864);
  _Float16* pWv  = (_Float16*)(ws + 32768 + 1572864 + 1179648);
  _Float16* pWo  = (_Float16*)(ws + 32768 + 1572864 + 2 * 1179648);

  pack_w_kernel<<<(DY_ / 32) * NT_, 32, 0, stream>>>(Wy, pWy);
  pack_w_kernel<<<(D_ / 32) * NT_, 32, 0, stream>>>(Wk, pWk);
  pack_w_kernel<<<(D_ / 32) * NT_, 32, 0, stream>>>(Wv, pWv);
  pack_w_kernel<<<(D_ / 32) * NT_, 32, 0, stream>>>(Wo, pWo);
  q_kernel<<<B_, 256, 0, stream>>>(x, Wx, bx, Wq, bq, q_ws);
  attn_main_kernel<<<(B_ * NX_) / WPB_, 256, 0, stream>>>(
      y, by, pWy, pWk, pWv, pWo, bk, bv, bo, ln_g, ln_b, prompt, q_ws, out);
}
